// RegionProposalNet_10960756540413
// MI455X (gfx1250) — compile-verified, matrix operand order checked by reading, not run on hardware
//
#include <hip/hip_runtime.h>
#include <math.h>

typedef __attribute__((ext_vector_type(16))) _Float16 v16h;
typedef __attribute__((ext_vector_type(8)))  float    v8f;
typedef __attribute__((ext_vector_type(4)))  unsigned u32x4;
typedef __attribute__((ext_vector_type(8)))  int      i32x8;
typedef __attribute__((ext_vector_type(4)))  int      i32x4;

union AFrag { v16h v; unsigned u[8]; };

#define BATCH   8
#define HH      50
#define WW      50
#define CIN     512
#define CMID    512
#define MTOT    (BATCH*HH*WW)      // 20000 output pixels (all batches)
#define NANCH   22500              // anchors per batch (2500*9)
#define NPRE    1000
#define NPOST   500
#define SORTN   32768              // pow2 >= 22500

// output float offsets
#define OFF_LOC   0
#define OFF_SCORE 720000
#define OFF_ROIS  1080000
#define OFF_RIDX  1096000
#define OFF_ANC   1100000

// K-index pattern for 16-bit WMMA fragments (doc: 16-bit A 16x32 layout).
__device__ __forceinline__ int frag_k(int p, int half) {
  return ((p & 4) << 2) + (half << 3) + ((p & 3) << 1);
}

// ---------------- prep: conv weights -> f16 [tap][k][n] ----------------
__global__ void prep_conv_w(const float* __restrict__ w, _Float16* __restrict__ bw)
{
  int idx = blockIdx.x * 256 + threadIdx.x;
  if (idx >= 9 * 512 * 512) return;
  int n   = idx & 511;
  int k   = (idx >> 9) & 511;
  int tap = idx >> 18;
  bw[idx] = (_Float16)w[(n * 512 + k) * 9 + tap];   // conv1_w [O][I][3][3]
}

// ---------------- prep: head weights -> f16 [n(64)][k(512)], bias[64] ----------------
__global__ void prep_heads(const float* __restrict__ loc_w, const float* __restrict__ score_w,
                           const float* __restrict__ loc_b, const float* __restrict__ score_b,
                           _Float16* __restrict__ bh, float* __restrict__ hb)
{
  int idx = blockIdx.x * 256 + threadIdx.x;
  if (idx >= 64 * 512) return;
  int n = idx >> 9, k = idx & 511;
  float v = 0.f;
  if (n < 36) v = loc_w[n * 512 + k];
  else if (n < 54) v = score_w[(n - 36) * 512 + k];
  bh[idx] = (_Float16)v;
  if (k == 0) hb[n] = (n < 36) ? loc_b[n] : (n < 54 ? score_b[n - 36] : 0.f);
}

// ---------------- 3x3 conv as 9 accumulated implicit GEMMs (f16 WMMA, f32 acc) -------
// grid (ceil(20000/64), 4), block 256 (8 wave32). Tile: M=64 pixels x N=128 channels.
__global__ __launch_bounds__(256) void conv3x3_wmma(
    const float* __restrict__ x, const _Float16* __restrict__ bw,
    const float* __restrict__ bias, _Float16* __restrict__ feat)
{
  __shared__ unsigned Alds_u[64 * 17];    // 64 rows x 32 f16, padded stride 17 dwords
  __shared__ unsigned Blds_u[128 * 17];   // 128 n   x 32 f16 (transposed), stride 17
  _Float16* Alds = (_Float16*)Alds_u;
  _Float16* Blds = (_Float16*)Blds_u;

  const int t    = threadIdx.x;
  const int wid  = t >> 5, lane = t & 31;
  const int wm   = wid & 3, wn = wid >> 2;        // 4 waves along M, 2 along N
  const int half = lane >> 4, l16 = lane & 15;
  const int mblk = blockIdx.x * 64, nblk = blockIdx.y * 128;

  v8f acc[4] = {};

  for (int tap = 0; tap < 9; ++tap) {
    const int dy = tap / 3 - 1, dx = tap % 3 - 1;
    for (int k0 = 0; k0 < 512; k0 += 32) {
      if (k0 + 32 < 512)
        __builtin_prefetch((const void*)(bw + (tap * 512 + k0 + 32) * 512 + nblk), 0, 1);
      __syncthreads();   // previous iteration's LDS reads done
      // stage A: 64 pixels x 32 input channels (f32 -> f16), zero-padded borders
      #pragma unroll
      for (int i = 0; i < 8; ++i) {
        int idx = t + i * 256;
        int r = idx & 63, c = idx >> 6;
        int m = mblk + r;
        float v = 0.f;
        if (m < MTOT) {
          int b = m / 2500, pix = m % 2500;
          int y = pix / 50 + dy, xx = pix % 50 + dx;
          if (y >= 0 && y < 50 && xx >= 0 && xx < 50)
            v = x[((b * 512 + (k0 + c)) * 50 + y) * 50 + xx];
        }
        Alds[r * 34 + c] = (_Float16)v;
      }
      // stage B transposed: Blds[n][k]
      #pragma unroll
      for (int i = 0; i < 16; ++i) {
        int idx = t + i * 256;
        int n = idx & 127, k = idx >> 7;
        Blds[n * 34 + k] = bw[(tap * 512 + k0 + k) * 512 + nblk + n];
      }
      __syncthreads();
      // fragments + WMMA
      AFrag a;
      const int am = wm * 16 + l16;
      #pragma unroll
      for (int p = 0; p < 8; ++p) a.u[p] = Alds_u[am * 17 + (frag_k(p, half) >> 1)];
      #pragma unroll
      for (int nt = 0; nt < 4; ++nt) {
        AFrag bf;
        const int bn = wn * 64 + nt * 16 + l16;
        #pragma unroll
        for (int p = 0; p < 8; ++p) bf.u[p] = Blds_u[bn * 17 + (frag_k(p, half) >> 1)];
        acc[nt] = __builtin_amdgcn_wmma_f32_16x16x32_f16(
            false, a.v, false, bf.v, (short)0, acc[nt], false, false);
      }
    }
  }
  // epilogue: bias + relu -> feat (channel-last f16 [20000][512])
  for (int nt = 0; nt < 4; ++nt) {
    const int n = nblk + wn * 64 + nt * 16 + l16;
    const float bv = bias[n];
    #pragma unroll
    for (int r = 0; r < 8; ++r) {
      const int m = mblk + wm * 16 + (half << 3) + r;
      if (m < MTOT) {
        float v = acc[nt][r] + bv;
        feat[m * 512 + n] = (_Float16)(v > 0.f ? v : 0.f);
      }
    }
  }
}

// ---------------- heads GEMM: [20000 x 512] x [512 x 64] (loc36|score18|pad) ---------
// A (128x512 f16) and B (64x512 f16) staged whole-tile into LDS by the Tensor Data
// Mover (TENSOR_LOAD_TO_LDS), with D# pad fields inserting 1 dword per 256 dwords so
// each row has a 257-dword stride (257 % 64 == 1 -> conflict-free fragment reads).
__global__ __launch_bounds__(256) void heads_wmma(
    const _Float16* __restrict__ feat, const _Float16* __restrict__ bh,
    const float* __restrict__ hb, float* __restrict__ out)
{
  __shared__ __align__(16) unsigned Au_l[128 * 257];   // 131,584 B
  __shared__ __align__(16) unsigned Bu_l[64 * 257];    //  65,792 B  (total 197 KB < 320 KB)
  const int t = threadIdx.x;
  const int wid = t >> 5, lane = t & 31, half = lane >> 4, l16 = lane & 15;
  const int m0blk = blockIdx.x * 128;
  const int m0 = m0blk + wid * 16;

#if __has_builtin(__builtin_amdgcn_tensor_load_to_lds) && __has_builtin(__builtin_amdgcn_s_wait_tensorcnt)
  if (wid == 0) {   // one wave issues both TDM descriptors (EXEC ignored by TDM)
    unsigned ldsA = (unsigned)(size_t)(void*)Au_l;
    unsigned ldsB = (unsigned)(size_t)(void*)Bu_l;
    unsigned long long gaA = (unsigned long long)(size_t)(feat + (size_t)m0blk * 512);
    unsigned long long gaB = (unsigned long long)(size_t)bh;
    // D# group1 dword0: data_size=1 (2B), pad_enable, pad_interval=7 (256 dwords),
    // pad_amount=0 (1 dword)
    const int d0 = (1 << 16) | (1 << 20) | (7 << 22);
    u32x4 g0a, g0b;
    i32x8 g1a, g1b;
    i32x4 z4 = {0, 0, 0, 0};
    i32x8 z8 = {0, 0, 0, 0, 0, 0, 0, 0};
    g0a[0] = 1u; g0a[1] = ldsA; g0a[2] = (unsigned)gaA;
    g0a[3] = (unsigned)((gaA >> 32) & 0x01FFFFFFull) | (2u << 30);     // type=2 (image)
    g0b[0] = 1u; g0b[1] = ldsB; g0b[2] = (unsigned)gaB;
    g0b[3] = (unsigned)((gaB >> 32) & 0x01FFFFFFull) | (2u << 30);
    g1a[0] = d0;
    g1a[1] = (512 & 0xFFFF) << 16;                   // tensor_dim0 = 512 (lo16 in [31:16])
    g1a[2] = (MTOT & 0xFFFF) << 16;                  // dim0 hi16 | tensor_dim1 lo16
    g1a[3] = ((MTOT >> 16) & 0xFFFF) | (512 << 16);  // tensor_dim1 hi16 | tile_dim0 = 512
    g1a[4] = 128;                                    // tile_dim1 = 128 rows (OOB rows -> 0)
    g1a[5] = 512; g1a[6] = 0; g1a[7] = 0;            // tensor_dim0_stride = 512
    g1b[0] = d0;
    g1b[1] = (512 & 0xFFFF) << 16;
    g1b[2] = (64 & 0xFFFF) << 16;                    // tensor_dim1 = 64
    g1b[3] = (512 << 16);                            // tile_dim0 = 512
    g1b[4] = 64;                                     // tile_dim1 = 64 rows
    g1b[5] = 512; g1b[6] = 0; g1b[7] = 0;
    // clang-23 / therock toolchain: 6-arg form (g0, g1, g2, g3, extra, cpol)
    __builtin_amdgcn_tensor_load_to_lds(g0a, g1a, z4, z4, z8, 0);
    __builtin_amdgcn_tensor_load_to_lds(g0b, g1b, z4, z4, z8, 0);
    __builtin_amdgcn_s_wait_tensorcnt((short)0);
  }
  __syncthreads();
#else
  {
    const unsigned* Fu = (const unsigned*)feat;
    const unsigned* Bg = (const unsigned*)bh;
    #pragma unroll 4
    for (int i = 0; i < 128; ++i) {
      int idx = t + i * 256;
      int r = idx >> 8, c = idx & 255;
      int m = m0blk + r;
      Au_l[r * 257 + c] = (m < MTOT) ? Fu[m * 256 + c] : 0u;
    }
    #pragma unroll 4
    for (int i = 0; i < 64; ++i) {
      int idx = t + i * 256;
      int r = idx >> 8, c = idx & 255;
      Bu_l[r * 257 + c] = Bg[r * 256 + c];
    }
    __syncthreads();
  }
#endif

  v8f acc[4] = {};
  const int arow = (wid * 16 + l16) * 257;
  for (int k0 = 0; k0 < 512; k0 += 32) {
    AFrag a;
    #pragma unroll
    for (int p = 0; p < 8; ++p) a.u[p] = Au_l[arow + ((k0 + frag_k(p, half)) >> 1)];
    #pragma unroll
    for (int nt = 0; nt < 4; ++nt) {
      AFrag bf;
      const int bn = nt * 16 + l16;
      #pragma unroll
      for (int p = 0; p < 8; ++p) bf.u[p] = Bu_l[bn * 257 + ((k0 + frag_k(p, half)) >> 1)];
      acc[nt] = __builtin_amdgcn_wmma_f32_16x16x32_f16(
          false, a.v, false, bf.v, (short)0, acc[nt], false, false);
    }
  }
  for (int nt = 0; nt < 4; ++nt) {
    const int n = nt * 16 + l16;
    const float bv = hb[n];
    #pragma unroll
    for (int r = 0; r < 8; ++r) {
      const int m = m0 + (half << 3) + r;
      if (m < MTOT) {
        float v = acc[nt][r] + bv;
        if (n < 36)      out[OFF_LOC + m * 36 + n] = v;
        else if (n < 54) out[OFF_SCORE + m * 18 + (n - 36)] = v;
      }
    }
  }
}

// ---------------- softmax (pairwise) -> positive prob ----------------
__global__ void softmax_pos(const float* __restrict__ out, float* __restrict__ pos)
{
  int idx = blockIdx.x * 256 + threadIdx.x;
  if (idx >= BATCH * NANCH) return;
  float s0 = out[OFF_SCORE + idx * 2 + 0];
  float s1 = out[OFF_SCORE + idx * 2 + 1];
  pos[idx] = 1.f / (1.f + expf(s0 - s1));
}

// ---------------- decode boxes, clip, mask, pack sort keys ----------------
__global__ void make_boxes(const float* __restrict__ out, const float* __restrict__ pos,
                           const float* __restrict__ anchor, float* __restrict__ boxes,
                           unsigned long long* __restrict__ kv)
{
  int idx = blockIdx.x * 256 + threadIdx.x;
  if (idx >= BATCH * SORTN) return;
  int b = idx >> 15, i = idx & (SORTN - 1);
  if (i < NANCH) {
    const float* a = anchor + i * 4;
    const float* l = out + OFF_LOC + b * 90000 + i * 4;
    float w  = a[2] - a[0], h = a[3] - a[1];
    float cx = a[0] + 0.5f * w, cy = a[1] + 0.5f * h;
    float ncx = l[0] * w + cx, ncy = l[1] * h + cy;
    float nw  = expf(l[2]) * w, nh = expf(l[3]) * h;
    float x1 = fminf(fmaxf(ncx - 0.5f * nw, 0.f), 800.f);
    float y1 = fminf(fmaxf(ncy - 0.5f * nh, 0.f), 800.f);
    float x2 = fminf(fmaxf(ncx + 0.5f * nw, 0.f), 800.f);
    float y2 = fminf(fmaxf(ncy + 0.5f * nh, 0.f), 800.f);
    int o = (b * NANCH + i) * 4;
    boxes[o] = x1; boxes[o + 1] = y1; boxes[o + 2] = x2; boxes[o + 3] = y2;
    bool valid = (x2 - x1 >= 16.f) && (y2 - y1 >= 16.f);
    float s = valid ? pos[b * NANCH + i] : -INFINITY;
    unsigned sb  = __float_as_uint(s);
    unsigned ord = (sb & 0x80000000u) ? ~sb : (sb | 0x80000000u); // ascending-float map
    kv[idx] = ((unsigned long long)(~ord) << 32) | (unsigned)i;   // asc sort = desc score
  } else {
    kv[idx] = ((unsigned long long)0xFF800000u << 32) | (unsigned)i; // = -inf score
  }
}

// ---------------- per-batch full bitonic sort in 256KB LDS, emit top-1000 ------------
__global__ __launch_bounds__(1024) void sort_select(
    const unsigned long long* __restrict__ kv_in, const float* __restrict__ boxes,
    float* __restrict__ topbox, unsigned* __restrict__ topvalid)
{
  __shared__ unsigned long long kv[SORTN];   // 256 KB — CDNA5 allows 320 KB / workgroup
  const int b = blockIdx.x, t = threadIdx.x;
  for (int i = t; i < SORTN; i += 1024) kv[i] = kv_in[b * SORTN + i];
  __syncthreads();
  for (unsigned k = 2; k <= SORTN; k <<= 1) {
    for (unsigned j = k >> 1; j > 0; j >>= 1) {
      #pragma unroll 1
      for (unsigned s = 0; s < SORTN / 2 / 1024; ++s) {
        unsigned t2 = (unsigned)t + s * 1024;
        unsigned i  = ((t2 & ~(j - 1)) << 1) | (t2 & (j - 1));
        unsigned p  = i | j;
        bool up = ((i & k) == 0);
        unsigned long long a = kv[i], c = kv[p];
        if ((a > c) == up) { kv[i] = c; kv[p] = a; }
      }
      __syncthreads();
    }
  }
  for (int i = t; i < NPRE; i += 1024) {
    unsigned long long v = kv[i];
    unsigned idx = (unsigned)v;
    unsigned hi  = (unsigned)(v >> 32);
    unsigned valid = (hi != 0xFF800000u) ? 1u : 0u;
    unsigned g = valid ? idx : 0u;
    const float* bp = boxes + ((size_t)b * NANCH + g) * 4;
    int o = (b * NPRE + i) * 4;
    topbox[o] = bp[0]; topbox[o + 1] = bp[1]; topbox[o + 2] = bp[2]; topbox[o + 3] = bp[3];
    topvalid[b * NPRE + i] = valid;
  }
}

// ---------------- per-batch greedy NMS (bitmask in LDS) + scatter rois ---------------
__global__ __launch_bounds__(256) void nms_scatter(
    const float* __restrict__ topbox, const unsigned* __restrict__ topvalid,
    float* __restrict__ out)
{
  __shared__ unsigned mask[NPRE * 32];   // 128 KB: bit j of word w => iou(i, w*32+j)>0.7
  __shared__ float    bx[NPRE * 4];      // 16 KB
  __shared__ int      dest[NPRE];        // 4 KB
  const int b = blockIdx.x, t = threadIdx.x;
  for (int i = t; i < NPRE * 4; i += 256) bx[i] = topbox[b * NPRE * 4 + i];
  __syncthreads();
  for (int w = t; w < NPRE * 32; w += 256) {
    int i = w >> 5, wj = w & 31;
    float x1 = bx[i*4], y1 = bx[i*4+1], x2 = bx[i*4+2], y2 = bx[i*4+3];
    float ai = (x2 - x1) * (y2 - y1);
    unsigned bits = 0;
    for (int jj = 0; jj < 32; ++jj) {
      int j = (wj << 5) + jj;
      if (j < NPRE) {
        float u1 = fmaxf(x1, bx[j*4]),   v1 = fmaxf(y1, bx[j*4+1]);
        float u2 = fminf(x2, bx[j*4+2]), v2 = fminf(y2, bx[j*4+3]);
        float inter = fmaxf(u2 - u1, 0.f) * fmaxf(v2 - v1, 0.f);
        float aj = (bx[j*4+2] - bx[j*4]) * (bx[j*4+3] - bx[j*4+1]);
        float iou = inter / (ai + aj - inter + 1e-10f);
        if (iou > 0.7f) bits |= 1u << jj;
      }
    }
    mask[i * 32 + wj] = bits;
  }
  __syncthreads();
  if (t < 32) {                          // one full wave32: serial greedy pass
    unsigned kept = 0;                   // lane t owns bitmask word t
    int cnt = 0;
    for (int i = 0; i < NPRE; ++i) {
      unsigned m = mask[i * 32 + t] & kept;
      bool sup = __any(m != 0);
      bool vm  = topvalid[b * NPRE + i] != 0;
      bool ki  = vm && !sup;
      if (ki && t == (i >> 5)) kept |= 1u << (i & 31);
      if (t == 0) { dest[i] = ki ? cnt : -1; cnt += ki ? 1 : 0; }
    }
  }
  __syncthreads();
  for (int i = t; i < NPOST * 4; i += 256) out[OFF_ROIS + b * NPOST * 4 + i] = 0.f;
  for (int i = t; i < NPOST;     i += 256) out[OFF_RIDX + b * NPOST + i] = (float)b;
  __syncthreads();
  for (int i = t; i < NPRE; i += 256) {
    int d = dest[i];
    if (d >= 0 && d < NPOST) {
      int o = OFF_ROIS + (b * NPOST + d) * 4;
      out[o] = bx[i*4]; out[o+1] = bx[i*4+1]; out[o+2] = bx[i*4+2]; out[o+3] = bx[i*4+3];
    }
  }
}

// ---------------- anchor passthrough ----------------
__global__ void copy_anchor(const float* __restrict__ anchor, float* __restrict__ out)
{
  int i = blockIdx.x * 256 + threadIdx.x;
  if (i < NANCH * 4) out[OFF_ANC + i] = anchor[i];
}

extern "C" void kernel_launch(void* const* d_in, const int* in_sizes, int n_in,
                              void* d_out, int out_size, void* d_ws, size_t ws_size,
                              hipStream_t stream)
{
  (void)in_sizes; (void)n_in; (void)out_size; (void)ws_size;
  const float* x   = (const float*)d_in[0];
  const float* cw  = (const float*)d_in[1];
  const float* cb  = (const float*)d_in[2];
  const float* sw  = (const float*)d_in[3];
  const float* sb  = (const float*)d_in[4];
  const float* lw  = (const float*)d_in[5];
  const float* lb  = (const float*)d_in[6];
  const float* anc = (const float*)d_in[7];
  float* out = (float*)d_out;

  char* ws = (char*)d_ws;                                   // ~31.2 MB used
  _Float16* feat = (_Float16*)(ws);                         // 20,480,000 B
  _Float16* bwc  = (_Float16*)(ws + 20480000);              //  4,718,592 B
  _Float16* bh   = (_Float16*)(ws + 25198592);              //     65,536 B
  float* hb      = (float*)   (ws + 25264128);              //        256 B
  float* pos     = (float*)   (ws + 25264384);              //    720,000 B
  float* boxes   = (float*)   (ws + 25984384);              //  2,880,000 B
  unsigned long long* kv = (unsigned long long*)(ws + 28864384); // 2,097,152 B
  float* topbox  = (float*)   (ws + 30961536);              //    128,000 B
  unsigned* topvalid = (unsigned*)(ws + 31089536);          //     32,000 B

  prep_conv_w<<<(9 * 512 * 512 + 255) / 256, 256, 0, stream>>>(cw, bwc);
  prep_heads <<<(64 * 512 + 255) / 256, 256, 0, stream>>>(lw, sw, lb, sb, bh, hb);
  conv3x3_wmma<<<dim3((MTOT + 63) / 64, CMID / 128), 256, 0, stream>>>(x, bwc, cb, feat);
  heads_wmma <<<(MTOT + 127) / 128, 256, 0, stream>>>(feat, bh, hb, out);
  softmax_pos<<<(BATCH * NANCH + 255) / 256, 256, 0, stream>>>(out, pos);
  make_boxes <<<(BATCH * SORTN) / 256, 256, 0, stream>>>(out, pos, anc, boxes, kv);
  sort_select<<<BATCH, 1024, 0, stream>>>(kv, boxes, topbox, topvalid);
  nms_scatter<<<BATCH, 256, 0, stream>>>(topbox, topvalid, out);
  copy_anchor<<<(NANCH * 4 + 255) / 256, 256, 0, stream>>>(anc, out);
}
